// NTMDecoder_29652454212343
// MI455X (gfx1250) — compile-verified, matrix operand order checked by reading, not run on hardware
//
#include <hip/hip_runtime.h>
#include <hip/hip_bf16.h>

// ---------------- problem constants (from reference) ----------------
#define BB       32
#define TT       128
#define VV       32000
#define D_EMB    512
#define D_IN     256
#define D_SENT   512
#define D_OUT    32000
#define N_SLOTS  128
#define M_W      64
#define HH       512
#define PER_HEAD 70            // 64 key + beta + g + 3 shift + gamma
#define HEAD_DIM 268           // 2*PER_HEAD + 2*M_W
#define MPITCH   65            // LDS pitch for M rows (bank-conflict free)
#define EPSF     1e-8f

typedef __attribute__((ext_vector_type(16))) _Float16 v16h;
typedef __attribute__((ext_vector_type(8)))  _Float16 v8h;
typedef __attribute__((ext_vector_type(8)))  float    v8f;

// ---------------- helpers ----------------
__device__ __forceinline__ float softplusf(float x) {
    return (x > 20.f) ? x : log1pf(expf(x));
}

// deterministic hash-based Bernoulli(keep=0.8) mask, scaled by 1/0.8
__device__ __forceinline__ float drop_mask(unsigned idx) {
    unsigned x = idx * 2654435761u + 0x9E3779B9u;
    x ^= x >> 16; x *= 0x7feb352du;
    x ^= x >> 15; x *= 0x846ca68bu;
    x ^= x >> 16;
    return (x > 858993459u) ? 1.25f : 0.0f;   // P(drop)=0.2
}

// ======================================================================
// Kernel 1: gather embedding rows and convert to f16 (A of GEMM1)
// ======================================================================
__global__ void gather_emb_kernel(const int* __restrict__ tgt,
                                  const float* __restrict__ emb,
                                  _Float16* __restrict__ A) {
    const int row = blockIdx.x;                      // 0 .. B*T-1  (= b*T + t)
    const int v = tgt[row];
    const float* src = emb + (size_t)v * D_EMB;
    _Float16* dst = A + (size_t)row * D_EMB;
    for (int i = threadIdx.x; i < D_EMB; i += blockDim.x)
        dst[i] = (_Float16)src[i];
}

// ======================================================================
// Kernel 2: pack fp32 row-major weight [K x N] into f16 WMMA-B fragment
// order. Tile = 32(K) x 16(N). Within a tile, halves are stored lane-major
// so each lane loads its whole 16-half fragment with ONE 32B load.
//   lane (0..31): n = nt*16 + (lane&15); hi = lane>>4
//   half j<8 : k = kt*32 +      hi*8 + j
//   half j>=8: k = kt*32 + 16 + hi*8 + (j-8)
// Source W is read exactly once -> non-temporal load (keep L2 for Bp).
// ======================================================================
__global__ void pack_b_kernel(const float* __restrict__ W,
                              _Float16* __restrict__ Bp,
                              int K, int N) {
    const int idx = blockIdx.x * blockDim.x + threadIdx.x;
    const int total = K * N;
    if (idx >= total) return;
    const int KT = K >> 5;
    const int tile = idx >> 9;          // /512 halves per tile
    const int within = idx & 511;
    const int lane = within >> 4;
    const int j = within & 15;
    const int nt = tile / KT;
    const int kt = tile - nt * KT;
    const int n = nt * 16 + (lane & 15);
    const int hi = lane >> 4;
    const int k = kt * 32 + ((j < 8) ? (hi * 8 + j) : (16 + hi * 8 + (j - 8)));
    const float w = __builtin_nontemporal_load(&W[(size_t)k * N + n]);
    Bp[idx] = (_Float16)w;
}

// ======================================================================
// Kernel 3: WMMA GEMM  C[M,N] = A[M,K](f16,row-major) * Bp(packed f16) + bias
// Block: 256 threads = 8 waves as 2(M) x 4(N); wave tile 16x64 (4 accums).
// Requires M%32==0, N%256==0, K%32==0.
// ntStore=1: C written with non-temporal hint (write-once 524MB stream of
// the final GEMM must not evict the 32MB packed W_out held in L2).
// ======================================================================
__global__ __launch_bounds__(256) void wmma_gemm_kernel(
    const _Float16* __restrict__ A, const _Float16* __restrict__ Bp,
    const float* __restrict__ bias, float* __restrict__ C,
    int M, int N, int K, int ntStore) {
    const int tid  = threadIdx.x;
    const int lane = tid & 31;
    const int wave = tid >> 5;
    const int wm = wave & 1;
    const int wn = wave >> 1;
    const int rowBase = blockIdx.y * 32 + wm * 16;
    const int colBase = blockIdx.x * 256 + wn * 64;
    const int r  = lane & 15;
    const int hi = lane >> 4;
    const int KT = K >> 5;

    const _Float16* arow  = A  + (size_t)(rowBase + r) * K + hi * 8;
    const _Float16* bbase = Bp + ((size_t)(colBase >> 4) * KT) * 512 + lane * 16;

    v8f acc[4] = {};

    for (int kt = 0; kt < KT; ++kt) {
        v8h alo = *(const v8h*)(arow + kt * 32);        // K: {0..7}+hi*8
        v8h ahi = *(const v8h*)(arow + kt * 32 + 16);   // K: {16..23}+hi*8
        v16h a = __builtin_shufflevector(alo, ahi,
                    0,1,2,3,4,5,6,7,8,9,10,11,12,13,14,15);
#pragma unroll
        for (int j = 0; j < 4; ++j) {
            v16h bfrag = *(const v16h*)(bbase + ((size_t)j * KT + kt) * 512);
            acc[j] = __builtin_amdgcn_wmma_f32_16x16x32_f16(
                         false, a, false, bfrag, (short)0, acc[j], false, false);
        }
    }

#pragma unroll
    for (int j = 0; j < 4; ++j) {
        const int col = colBase + j * 16 + r;
        const float bv = bias ? bias[col] : 0.f;
#pragma unroll
        for (int v = 0; v < 8; ++v) {
            const int rw = rowBase + v + 8 * hi;        // C/D layout: row=v+8*hi
            float* cp = &C[(size_t)rw * N + col];
            const float val = acc[j][v] + bv;
            if (ntStore) __builtin_nontemporal_store(val, cp);
            else         *cp = val;
        }
    }
}

// ======================================================================
// NTM addressing (content + interpolate + shift + sharpen), one head.
// 128 slots; 512-thread block; wave-0 shuffle reductions (wave32).
// wio: in = w_prev, out = new w.
// ======================================================================
__device__ void address_head(float* Ms, const float* pp, int base, float* kvec,
                             float* wio, float* simb, float* wg, float* wt,
                             float* scal, int tid) {
    if (tid < M_W) kvec[tid] = tanhf(pp[base + tid]);
    if (tid == 64) scal[0] = softplusf(pp[base + 64]);                 // beta
    if (tid == 65) scal[1] = 1.f / (1.f + expf(-pp[base + 65]));       // g
    if (tid == 66) {                                                   // shift softmax
        float a0 = pp[base + 66], a1 = pp[base + 67], a2 = pp[base + 68];
        float m = fmaxf(a0, fmaxf(a1, a2));
        float e0 = expf(a0 - m), e1 = expf(a1 - m), e2 = expf(a2 - m);
        float s = e0 + e1 + e2;
        scal[2] = e0 / s; scal[3] = e1 / s; scal[4] = e2 / s;
    }
    if (tid == 67) scal[5] = 1.f + softplusf(pp[base + 69]);           // gamma
    __syncthreads();
    if (tid < 32) {                                                    // ||k||
        float v = kvec[tid] * kvec[tid] + kvec[tid + 32] * kvec[tid + 32];
#pragma unroll
        for (int off = 16; off; off >>= 1) v += __shfl_xor(v, off);
        if (tid == 0) scal[6] = sqrtf(v);
    }
    __syncthreads();
    if (tid < N_SLOTS) {                                               // cosine sim
        float dot = 0.f, nn = 0.f;
        for (int i = 0; i < M_W; ++i) {
            float m = Ms[tid * MPITCH + i];
            dot = fmaf(m, kvec[i], dot);
            nn  = fmaf(m, m, nn);
        }
        simb[tid] = dot / (sqrtf(nn) * scal[6] + EPSF);
    }
    __syncthreads();
    if (tid < 32) {                                                    // max(beta*sim)
        float v = fmaxf(fmaxf(simb[tid], simb[tid + 32]),
                        fmaxf(simb[tid + 64], simb[tid + 96]));
#pragma unroll
        for (int off = 16; off; off >>= 1) v = fmaxf(v, __shfl_xor(v, off));
        if (tid == 0) scal[7] = scal[0] * v;
    }
    __syncthreads();
    if (tid < N_SLOTS) simb[tid] = expf(scal[0] * simb[tid] - scal[7]);
    __syncthreads();
    if (tid < 32) {
        float v = simb[tid] + simb[tid + 32] + simb[tid + 64] + simb[tid + 96];
#pragma unroll
        for (int off = 16; off; off >>= 1) v += __shfl_xor(v, off);
        if (tid == 0) scal[8] = v;
    }
    __syncthreads();
    if (tid < N_SLOTS) {                                               // interpolate
        float wc = simb[tid] / scal[8];
        wg[tid] = scal[1] * wc + (1.f - scal[1]) * wio[tid];
    }
    __syncthreads();
    if (tid < N_SLOTS) {                                               // shift + sharpen
        float v = scal[2] * wg[(tid + 1) & 127] + scal[3] * wg[tid]
                + scal[4] * wg[(tid + 127) & 127];
        wt[tid] = __powf(v + EPSF, scal[5]);
    }
    __syncthreads();
    if (tid < 32) {
        float v = wt[tid] + wt[tid + 32] + wt[tid + 64] + wt[tid + 96];
#pragma unroll
        for (int off = 16; off; off >>= 1) v += __shfl_xor(v, off);
        if (tid == 0) scal[9] = v;
    }
    __syncthreads();
    if (tid < N_SLOTS) wio[tid] = wt[tid] / scal[9];
    __syncthreads();
}

// ======================================================================
// Kernel 4: NTM recurrence. 32 blocks (one per batch element) x 512 thr.
// M matrix lives in LDS (pitch 65 -> conflict-free column access).
// Emits post-ReLU, post-dropout o_t in f16 (A of the big GEMM).
// ======================================================================
__global__ __launch_bounds__(512) void ntm_scan_kernel(
    const float* __restrict__ X, const float* __restrict__ M0,
    const float* __restrict__ W_c,    const float* __restrict__ b_c,
    const float* __restrict__ W_head, const float* __restrict__ b_head,
    const float* __restrict__ W_o,    const float* __restrict__ b_o,
    _Float16* __restrict__ O) {
    __shared__ float Ms[N_SLOTS * MPITCH];
    __shared__ float rr[M_W];
    __shared__ float hh[HH];
    __shared__ float pp[HEAD_DIM];
    __shared__ float xs[D_IN];
    __shared__ float w_r[N_SLOTS], w_w[N_SLOTS];
    __shared__ float kvec[M_W], erasev[M_W], addv[M_W];
    __shared__ float simb[N_SLOTS], wg[N_SLOTS], wt[N_SLOTS];
    __shared__ float scal[16];

    const int b = blockIdx.x;
    const int tid = threadIdx.x;

    for (int i = tid; i < N_SLOTS * M_W; i += 512) {
        int n = i >> 6, c = i & 63;
        Ms[n * MPITCH + c] = M0[i];
    }
    if (tid < N_SLOTS) { w_r[tid] = 1.0f / N_SLOTS; w_w[tid] = 1.0f / N_SLOTS; }
    if (tid < M_W) rr[tid] = 0.f;
    __syncthreads();

    for (int t = 0; t < TT; ++t) {
        const int row = b * TT + t;
        if (tid < D_IN) xs[tid] = X[(size_t)row * D_IN + tid];
        __syncthreads();
        {   // h = tanh([x, r] @ W_c + b_c), one output per thread
            float acc = b_c[tid];
            for (int j = 0; j < D_IN; ++j) acc = fmaf(xs[j], W_c[j * HH + tid], acc);
            for (int j = 0; j < M_W;  ++j) acc = fmaf(rr[j], W_c[(D_IN + j) * HH + tid], acc);
            hh[tid] = tanhf(acc);
        }
        __syncthreads();
        if (tid < HEAD_DIM) {   // p = h @ W_head + b_head
            float acc = b_head[tid];
            for (int j = 0; j < HH; ++j) acc = fmaf(hh[j], W_head[j * HEAD_DIM + tid], acc);
            pp[tid] = acc;
        }
        __syncthreads();
        if (tid < M_W) {
            erasev[tid] = 1.f / (1.f + expf(-pp[2 * PER_HEAD + tid]));
            addv[tid]   = tanhf(pp[2 * PER_HEAD + M_W + tid]);
        }
        // write-head addressing (p[PER_HEAD : 2*PER_HEAD])
        address_head(Ms, pp, PER_HEAD, kvec, w_w, simb, wg, wt, scal, tid);
        // memory update
        for (int i = tid; i < N_SLOTS * M_W; i += 512) {
            int n = i >> 6, c = i & 63;
            float wv = w_w[n];
            float m = Ms[n * MPITCH + c];
            Ms[n * MPITCH + c] = m * (1.f - wv * erasev[c]) + wv * addv[c];
        }
        __syncthreads();
        // read-head addressing on updated M (p[0 : PER_HEAD])
        address_head(Ms, pp, 0, kvec, w_r, simb, wg, wt, scal, tid);
        if (tid < M_W) {        // r_n = w_r @ M
            float acc = 0.f;
            for (int n = 0; n < N_SLOTS; ++n)
                acc = fmaf(w_r[n], Ms[n * MPITCH + tid], acc);
            rr[tid] = acc;
        }
        __syncthreads();
        {   // o = relu([h, r_n] @ W_o + b_o) * mask  -> f16
            float acc = b_o[tid];
            for (int j = 0; j < HH;  ++j) acc = fmaf(hh[j], W_o[j * D_SENT + tid], acc);
            for (int j = 0; j < M_W; ++j) acc = fmaf(rr[j], W_o[(HH + j) * D_SENT + tid], acc);
            float o = fmaxf(acc, 0.f) * drop_mask(((unsigned)t * BB + b) * D_SENT + tid);
            O[(size_t)row * D_SENT + tid] = (_Float16)o;
        }
        __syncthreads();
    }
}

// ======================================================================
// launch
// ======================================================================
extern "C" void kernel_launch(void* const* d_in, const int* in_sizes, int n_in,
                              void* d_out, int out_size, void* d_ws, size_t ws_size,
                              hipStream_t stream) {
    (void)in_sizes; (void)n_in; (void)out_size; (void)ws_size;
    const int*   tgt    = (const int*)  d_in[0];
    const float* emb    = (const float*)d_in[4];
    const float* W_lin  = (const float*)d_in[5];
    const float* b_lin  = (const float*)d_in[6];
    const float* W_c    = (const float*)d_in[7];
    const float* b_c    = (const float*)d_in[8];
    const float* W_head = (const float*)d_in[9];
    const float* b_head = (const float*)d_in[10];
    const float* W_o    = (const float*)d_in[11];
    const float* b_o    = (const float*)d_in[12];
    const float* W_out  = (const float*)d_in[13];
    const float* b_out  = (const float*)d_in[14];
    const float* M0     = (const float*)d_in[15];
    float* out = (float*)d_out;

    char* ws = (char*)d_ws;
    size_t off = 0;
    auto alloc = [&](size_t bytes) -> void* {
        void* p = ws + off;
        off = (off + bytes + 255) & ~(size_t)255;
        return p;
    };
    const int ROWS = BB * TT;                                      // 4096
    _Float16* A1    = (_Float16*)alloc((size_t)ROWS * D_EMB * 2);  // gathered emb, f16
    _Float16* BpLin = (_Float16*)alloc((size_t)D_EMB * D_IN * 2);  // packed W_lin
    float*    X     = (float*)   alloc((size_t)ROWS * D_IN * 4);   // dec_input
    _Float16* Of    = (_Float16*)alloc((size_t)ROWS * D_SENT * 2); // masked relu out
    _Float16* BpOut = (_Float16*)alloc((size_t)D_SENT * D_OUT * 2);// packed W_out

    gather_emb_kernel<<<ROWS, 256, 0, stream>>>(tgt, emb, A1);
    pack_b_kernel<<<(D_EMB * D_IN + 255) / 256, 256, 0, stream>>>(W_lin, BpLin, D_EMB, D_IN);
    pack_b_kernel<<<(D_SENT * D_OUT + 255) / 256, 256, 0, stream>>>(W_out, BpOut, D_SENT, D_OUT);

    dim3 g1(D_IN / 256, ROWS / 32);        // (1, 128)
    wmma_gemm_kernel<<<g1, 256, 0, stream>>>(A1, BpLin, b_lin, X, ROWS, D_IN, D_EMB, 0);

    ntm_scan_kernel<<<BB, 512, 0, stream>>>(X, M0, W_c, b_c, W_head, b_head, W_o, b_o, Of);

    dim3 g2(D_OUT / 256, ROWS / 32);       // (125, 128)
    wmma_gemm_kernel<<<g2, 256, 0, stream>>>(Of, BpOut, b_out, out, ROWS, D_OUT, D_SENT, 1);
}
